// ARContextBlock_7619271983796
// MI455X (gfx1250) — matrix operand order repaired and checked by reading, not compiled, not run on hardware
//
#include <hip/hip_runtime.h>
#include <hip/hip_bf16.h>

typedef _Float16 f16;
typedef __attribute__((ext_vector_type(16))) _Float16 v16h;
typedef __attribute__((ext_vector_type(8)))  _Float16 v8h;
typedef __attribute__((ext_vector_type(4)))  _Float16 v4h;
typedef __attribute__((ext_vector_type(8)))  float    v8f;

#define WMMA_F16F32(A, B, C) \
  __builtin_amdgcn_wmma_f32_16x16x32_f16(false, (A), false, (B), (short)0, (C), false, false)

// ---------------------------------------------------------------------------
// A-operand (16x32 f16): lane l -> row = l&15 ; elems 0..7  : K = (l>>4)*8 + e
//                                              elems 8..15 : K = 16 + (l>>4)*8 + e
__device__ __forceinline__ v16h load_a_16x32(const f16* rowp, int lh) {
  v8h lo = *(const v8h*)(rowp + lh * 8);
  v8h hi = *(const v8h*)(rowp + lh * 8 + 16);
  v16h r;
#pragma unroll
  for (int e = 0; e < 8; ++e) { r[e] = lo[e]; r[e + 8] = hi[e]; }
  return r;
}

// B-operand (32x16 f16): lane l -> col = l&15 ; elems 0..7 : K = (l>>4)*8 + e ; elems 8..15 : K = 16 + (l>>4)*8 + e
__device__ __forceinline__ v16h load_b_32x16(const f16* colp, int lh) {
  v8h lo = *(const v8h*)(colp + lh * 8);
  v8h hi = *(const v8h*)(colp + lh * 8 + 16);
  v16h r;
#pragma unroll
  for (int e = 0; e < 8; ++e) { r[e] = lo[e]; r[e + 8] = hi[e]; }
  return r;
}

// ---------------------------------------------------------------------------
__global__ void f32_to_f16_kernel(const float* __restrict__ src, f16* __restrict__ dst, int n) {
  int i = blockIdx.x * 256 + threadIdx.x;
  if (i < n) dst[i] = (f16)src[i];
}

// LayerNorm over C=128, one wave (32 lanes x 4 floats) per token, f16 output.
__global__ __launch_bounds__(256) void ln128_kernel(const float* __restrict__ x,
                                                    const float* __restrict__ w,
                                                    const float* __restrict__ b,
                                                    f16* __restrict__ y, int n_tokens) {
  int wid = threadIdx.x >> 5, l = threadIdx.x & 31;
  long tok = (long)blockIdx.x * 8 + wid;
  if (tok >= n_tokens) return;
  const float4 v  = ((const float4*)(x + tok * 128))[l];
  float s = v.x + v.y + v.z + v.w;
#pragma unroll
  for (int o = 16; o > 0; o >>= 1) s += __shfl_xor(s, o);
  float mu = s * (1.f / 128.f);
  float d0 = v.x - mu, d1 = v.y - mu, d2 = v.z - mu, d3 = v.w - mu;
  float q = d0 * d0 + d1 * d1 + d2 * d2 + d3 * d3;
#pragma unroll
  for (int o = 16; o > 0; o >>= 1) q += __shfl_xor(q, o);
  float rstd = rsqrtf(q * (1.f / 128.f) + 1e-5f);
  const float4 wv = ((const float4*)w)[l];
  const float4 bv = ((const float4*)b)[l];
  v4h r;
  r[0] = (f16)(d0 * rstd * wv.x + bv.x);
  r[1] = (f16)(d1 * rstd * wv.y + bv.y);
  r[2] = (f16)(d2 * rstd * wv.z + bv.z);
  r[3] = (f16)(d3 * rstd * wv.w + bv.w);
  *(v4h*)(y + tok * 128 + 4 * l) = r;
}

// ---------------------------------------------------------------------------
// WMMA GEMM: D[M,N] = A[M,K](f16) * W[N,K]^T(f16) + bias [+GELU] [+res]
// 4 waves / block; each wave computes a 32x64 macro-tile (2 A x 4 B operands,
// 8 WMMAs per 32-wide K step) for ~22 FLOP/B of L2 traffic.
// grid = (M/128, N/64). Requires M%128==0, N%64==0, K%32==0.
__global__ __launch_bounds__(128) void wmma_gemm_kernel(
    const f16* __restrict__ A, const f16* __restrict__ W, const float* __restrict__ bias,
    const float* __restrict__ residual, float* __restrict__ outf, f16* __restrict__ outh,
    int M, int N, int K, int act_gelu) {
  const int l = threadIdx.x & 31, ll = l & 15, lh = l >> 4;
  const int wid = threadIdx.x >> 5;
  const long m0 = ((long)blockIdx.x * 4 + wid) * 32;
  const long n0 = (long)blockIdx.y * 64;

  const f16* arow0 = A + (m0 + ll) * (long)K;
  const f16* arow1 = arow0 + 16 * (long)K;
  const f16* wrow0 = W + (n0 + ll) * (long)K;

  v8f acc[8];
#pragma unroll
  for (int i = 0; i < 8; ++i) acc[i] = (v8f){};

  for (int k0 = 0; k0 < K; k0 += 32) {
    if (k0 + 32 < K) {
      __builtin_prefetch(arow0 + k0 + 32, 0, 1);   // global_prefetch_b8
      __builtin_prefetch(arow1 + k0 + 32, 0, 1);
    }
    v16h a0 = load_a_16x32(arow0 + k0, lh);
    v16h a1 = load_a_16x32(arow1 + k0, lh);
    v16h bt[4];
#pragma unroll
    for (int j = 0; j < 4; ++j) bt[j] = load_b_32x16(wrow0 + (long)j * 16 * K + k0, lh);
#pragma unroll
    for (int j = 0; j < 4; ++j) {
      acc[j]     = WMMA_F16F32(a0, bt[j], acc[j]);
      acc[4 + j] = WMMA_F16F32(a1, bt[j], acc[4 + j]);
    }
  }

  float bn[4];
#pragma unroll
  for (int j = 0; j < 4; ++j) bn[j] = bias ? bias[n0 + j * 16 + ll] : 0.f;

#pragma unroll
  for (int i = 0; i < 2; ++i) {
#pragma unroll
    for (int j = 0; j < 4; ++j) {
#pragma unroll
      for (int v = 0; v < 8; ++v) {
        long row = m0 + i * 16 + v + 8 * lh;       // C/D layout: row = vgpr + 8*(lane>=16)
        long idx = row * N + n0 + j * 16 + ll;     // col = lane&15
        float val = acc[i * 4 + j][v] + bn[j];
        if (act_gelu) val = 0.5f * val * (1.f + erff(val * 0.70710678118f));
        if (residual) val += residual[idx];
        if (outf) outf[idx] = val;
        if (outh) outh[idx] = (f16)val;
      }
    }
  }
}

// ---------------------------------------------------------------------------
// Flash attention: 4 waves / block, one wave per 16-query tile. dh = 32.
// S^T trick: score tiles computed as K*Q^T so P packs directly into the WMMA
// A-operand layout for P*V (no LDS transpose, no cross-lane P movement).
// mask_mode 0: causal (k <= q). mask_mode 1: band (|k-q| <= band).
__global__ __launch_bounds__(128) void flash_attn_kernel(
    const f16* __restrict__ qkv, f16* __restrict__ out, int S, int stride_q,
    int qd1, long qm1, long qm2, int od1, long om1, long om2, int stride_o,
    int koff, int voff, int mask_mode, int band, float scale) {
  const int l = threadIdx.x & 31, ll = l & 15, lh = l >> 4;
  const int wid = threadIdx.x >> 5;
  const int q0 = (blockIdx.x * 4 + wid) * 16;
  const int h  = blockIdx.y;
  const int s  = blockIdx.z;
  const long qbase = (long)(s / qd1) * qm1 + (long)(s % qd1) * qm2;
  const long obase = (long)(s / od1) * om1 + (long)(s % od1) * om2;

  // B-operand = Q^T (32 x 16): col = query q0+ll, elem e -> dh = (e>=8)*16 + lh*8 + (e&7)
  const v16h Bq = load_b_32x16(qkv + qbase + (long)(q0 + ll) * stride_q + h * 32, lh);

  v8f acc0 = {}, acc1 = {};           // O tiles: dh 0..15 and 16..31
  float m_run = -1e30f, l_run = 0.f;
  const int q_glob = q0 + ll;

  int kbeg, kend;
  if (mask_mode == 0) { kbeg = 0; kend = q0 + 16; }
  else { int kb = q0 - band; kbeg = kb < 0 ? 0 : (kb & ~31); kend = q0 + 16 + band; }
  if (kend > S) kend = S;

  for (int kt = kbeg; kt < kend; kt += 32) {
    // --- two S^T = K * Q^T tiles (keys kt..kt+15, kt+16..kt+31) ---
    v8f s0 = {}, s1 = {};
    {
      int kr = kt + ll;
      v16h Ak = {};
      if (kr < S) Ak = load_a_16x32(qkv + qbase + (long)kr * stride_q + koff + h * 32, lh);
      s0 = WMMA_F16F32(Ak, Bq, s0);
    }
    {
      int kr = kt + 16 + ll;
      v16h Ak = {};
      if (kr < S) Ak = load_a_16x32(qkv + qbase + (long)kr * stride_q + koff + h * 32, lh);
      s1 = WMMA_F16F32(Ak, Bq, s1);
    }
    // S^T C/D layout: this lane holds query col ll, key rows v + 8*lh (+16 tile1)
    float x0[8], x1v[8], tmax = -1e30f;
#pragma unroll
    for (int v = 0; v < 8; ++v) {
      int k0i = kt + v + 8 * lh, k1i = k0i + 16;
      bool ok0 = (mask_mode == 0) ? (k0i <= q_glob) : (abs(k0i - q_glob) <= band);
      bool ok1 = (mask_mode == 0) ? (k1i <= q_glob) : (abs(k1i - q_glob) <= band);
      ok0 = ok0 && (k0i < S); ok1 = ok1 && (k1i < S);
      x0[v]  = ok0 ? s0[v] * scale : -1e30f;
      x1v[v] = ok1 ? s1[v] * scale : -1e30f;
      tmax = fmaxf(tmax, fmaxf(x0[v], x1v[v]));
    }
    tmax = fmaxf(tmax, __shfl_xor(tmax, 16));     // combine both key halves of query ll
    float m_new = fmaxf(m_run, tmax);
    float alpha = __expf(m_run - m_new);
    float ts = 0.f;
    v16h Ap;                                      // P in A-operand layout directly
#pragma unroll
    for (int v = 0; v < 8; ++v) {
      float p0 = (x0[v]  > -1e29f) ? __expf(x0[v]  - m_new) : 0.f;
      float p1 = (x1v[v] > -1e29f) ? __expf(x1v[v] - m_new) : 0.f;
      ts += p0 + p1;
      Ap[v] = (f16)p0; Ap[v + 8] = (f16)p1;
    }
    ts += __shfl_xor(ts, 16);
    l_run = l_run * alpha + ts;
    m_run = m_new;
    // rescale O accumulators: row = j + 8*lh -> alpha lives in lane (j + 8*lh)
#pragma unroll
    for (int j = 0; j < 8; ++j) {
      float aj = __shfl(alpha, j + 8 * lh);
      acc0[j] *= aj; acc1[j] *= aj;
    }
    // --- V B-operands: col = dh, elem e -> key kt + (e>=8)*16 + lh*8 + (e&7) ---
    v16h Bv0, Bv1;
    const f16* vp = qkv + qbase + voff + h * 32;
#pragma unroll
    for (int e = 0; e < 16; ++e) {
      int kr = kt + ((e & 8) ? 16 : 0) + lh * 8 + (e & 7);
      if (kr >= S) kr = S - 1;                    // safe: masked keys have p == 0
      Bv0[e] = vp[(long)kr * stride_q + ll];
      Bv1[e] = vp[(long)kr * stride_q + 16 + ll];
    }
    acc0 = WMMA_F16F32(Ap, Bv0, acc0);
    acc1 = WMMA_F16F32(Ap, Bv1, acc1);
  }

  // normalize + store: O row = query j + 8*lh, col = dh ll (+16)
#pragma unroll
  for (int j = 0; j < 8; ++j) {
    float lr = __shfl(l_run, j + 8 * lh);
    float inv = 1.f / lr;
    long row = q0 + j + 8 * lh;
    f16* op = out + obase + row * (long)stride_o + h * 32;
    op[ll]      = (f16)(acc0[j] * inv);
    op[16 + ll] = (f16)(acc1[j] * inv);
  }
}

// ---------------------------------------------------------------------------
extern "C" void kernel_launch(void* const* d_in, const int* in_sizes, int n_in,
                              void* d_out, int out_size, void* d_ws, size_t ws_size,
                              hipStream_t stream) {
  (void)in_sizes; (void)n_in; (void)out_size; (void)ws_size;
  constexpr int B = 2, T = 256, FB = 128, C = 128, H = 4, BAND = 16, INNER = 4 * C;
  constexpr long NTOK = (long)B * T * FB;          // 65536
  const float scale = 0.17677669529663687f;        // 1/sqrt(32)

  const float* x     = (const float*)d_in[0];
  const float* ln_tw = (const float*)d_in[1];  const float* ln_tb = (const float*)d_in[2];
  const float* t_wi  = (const float*)d_in[3];  const float* t_bi  = (const float*)d_in[4];
  const float* t_wo  = (const float*)d_in[5];  const float* t_bo  = (const float*)d_in[6];
  const float* ln_fw = (const float*)d_in[7];  const float* ln_fb = (const float*)d_in[8];
  const float* f_wi  = (const float*)d_in[9];  const float* f_bi  = (const float*)d_in[10];
  const float* f_wo  = (const float*)d_in[11]; const float* f_bo  = (const float*)d_in[12];
  const float* ln_nw = (const float*)d_in[13]; const float* ln_nb = (const float*)d_in[14];
  const float* w1    = (const float*)d_in[15]; const float* b1    = (const float*)d_in[16];
  const float* w2    = (const float*)d_in[17]; const float* b2    = (const float*)d_in[18];
  float* out = (float*)d_out;

  // workspace carve-up (256B aligned)
  char* w = (char*)d_ws;
  f16*   xln  = (f16*)w;                              w += NTOK * C * sizeof(f16);       // 16 MB
  f16*   big  = (f16*)w;                              w += NTOK * INNER * sizeof(f16);   // 64 MB (qkv / h)
  f16*   att  = (f16*)w;                              w += NTOK * C * sizeof(f16);       // 16 MB
  float* x1   = (float*)w;                            w += NTOK * C * sizeof(float);     // 33.5 MB
  f16*   wh   = (f16*)w;                                                                 // f16 weights
  f16 *t_wi_h = wh,            *t_wo_h = t_wi_h + 3*C*C, *f_wi_h = t_wo_h + C*C;
  f16 *f_wo_h = f_wi_h + 3*C*C, *w1_h  = f_wo_h + C*C,   *w2_h   = w1_h + INNER*C;

  auto cvt = [&](const float* s, f16* d, int n) {
    f32_to_f16_kernel<<<(n + 255) / 256, 256, 0, stream>>>(s, d, n);
  };
  cvt(t_wi, t_wi_h, 3*C*C); cvt(t_wo, t_wo_h, C*C);
  cvt(f_wi, f_wi_h, 3*C*C); cvt(f_wo, f_wo_h, C*C);
  cvt(w1, w1_h, INNER*C);   cvt(w2, w2_h, C*INNER);

  dim3 lnGrid((NTOK + 7) / 8), lnBlk(256), quadBlk(128);
  auto gemm = [&](const f16* A, const f16* W_, const float* bias, const float* res,
                  float* of, f16* oh, int M, int N, int K, int gelu) {
    wmma_gemm_kernel<<<dim3(M / 128, N / 64), quadBlk, 0, stream>>>(A, W_, bias, res, of, oh,
                                                                    M, N, K, gelu);
  };

  // ---- stage 1: temporal causal MHSA (sequences over T, fixed b,f) ----
  ln128_kernel<<<lnGrid, lnBlk, 0, stream>>>(x, ln_tw, ln_tb, xln, NTOK);
  gemm(xln, t_wi_h, t_bi, nullptr, nullptr, big, NTOK, 3 * C, C, 0);        // qkv
  flash_attn_kernel<<<dim3((T / 16) / 4, H, B * FB), quadBlk, 0, stream>>>(
      big, att, T, FB * 3 * C,
      /*q: s=(b,f)*/ FB, (long)T * FB * 3 * C, 3 * C,
      /*o:*/          FB, (long)T * FB * C,     C,    FB * C,
      C, 2 * C, /*causal*/ 0, 0, scale);
  gemm(att, t_wo_h, t_bo, x, x1, nullptr, NTOK, C, C, 0);                   // x1 = x + proj

  // ---- stage 2: banded frequency MHSA (sequences over F, fixed b,t) ----
  ln128_kernel<<<lnGrid, lnBlk, 0, stream>>>(x1, ln_fw, ln_fb, xln, NTOK);
  gemm(xln, f_wi_h, f_bi, nullptr, nullptr, big, NTOK, 3 * C, C, 0);
  flash_attn_kernel<<<dim3((FB / 16) / 4, H, B * T), quadBlk, 0, stream>>>(
      big, att, FB, 3 * C,
      /*q: s=(b,t)*/ B * T, 0L, (long)FB * 3 * C,
      /*o:*/         B * T, 0L, (long)FB * C,     C,
      C, 2 * C, /*band*/ 1, BAND, scale);
  gemm(att, f_wo_h, f_bo, x1, out, nullptr, NTOK, C, C, 0);                 // x2 -> d_out

  // ---- stage 3: FFN ----
  ln128_kernel<<<lnGrid, lnBlk, 0, stream>>>(out, ln_nw, ln_nb, xln, NTOK);
  gemm(xln, w1_h, b1, nullptr, nullptr, big, NTOK, INNER, C, /*gelu*/ 1);   // h
  gemm(big, w2_h, b2, out, out, nullptr, NTOK, C, INNER, 0);               // d_out = x2 + h@w2^T
}
